// MultiHeadAttention_49727131353242
// MI455X (gfx1250) — matrix operand order repaired and checked by reading, not compiled, not run on hardware
//
#include <hip/hip_runtime.h>

typedef __attribute__((ext_vector_type(16))) _Float16 v16h;
typedef __attribute__((ext_vector_type(8)))  _Float16 v8h;
typedef __attribute__((ext_vector_type(4)))  _Float16 v4h;
typedef __attribute__((ext_vector_type(8)))  float    v8f;

#define Bc 4
#define Sc 2048
#define Dc 1024
#define Hc 16
#define HDc 64
#define Mc (Bc * Sc)      // 8192 rows

#define BM 128
#define BN 128
#define BK 32
#define LDT 40            // padded LDS row stride (f16), %8==0
#define KB 128            // attention key-block
#define LDP 136           // P-buffer row stride (f16), 272B, %8==0

#if __has_builtin(__builtin_amdgcn_global_load_async_to_lds_b128)
#define HAS_ASYNC 1
typedef int v4i32 __attribute__((vector_size(16)));
typedef __attribute__((address_space(1))) v4i32 gv4i;  // global int4
typedef __attribute__((address_space(3))) v4i32 lv4i;  // LDS int4
#endif

__device__ __forceinline__ v8f wmma_f16(v16h a, v16h b, v8f c) {
  return __builtin_amdgcn_wmma_f32_16x16x32_f16(false, a, false, b, (short)0, c,
                                                false, false);
}

__device__ __forceinline__ v16h cat16(v8h lo, v8h hi) {
  return __builtin_shufflevector(lo, hi, 0, 1, 2, 3, 4, 5, 6, 7, 8, 9, 10, 11,
                                 12, 13, 14, 15);
}

// A fragment 16x32 f16 (ISA 7.12.2): lane m=L%16, half=L/16.
// elems 0..7 -> k=half*8+e ; elems 8..15 -> k=half*8+16+(e-8)
__device__ __forceinline__ v16h load_frag_a(const _Float16* p, int ld) {
  const int lane = threadIdx.x & 31;
  const _Float16* r = p + (lane & 15) * ld + (lane >> 4) * 8;
  return cat16(*(const v8h*)(r), *(const v8h*)(r + 16));
}

// B fragment 32x16 f16 from n-major storage q[n][k]: lane n=L%16,
// elems 0..7 -> k=half*8+e ; elems 8..15 -> k=half*8+16+(e-8)
__device__ __forceinline__ v16h load_frag_b(const _Float16* p, int ld) {
  const int lane = threadIdx.x & 31;
  const _Float16* r = p + (lane & 15) * ld + (lane >> 4) * 8;
  return cat16(*(const v8h*)(r), *(const v8h*)(r + 16));
}

__device__ __forceinline__ void copy16(_Float16* lds, const _Float16* g) {
#ifdef HAS_ASYNC
  __builtin_amdgcn_global_load_async_to_lds_b128(
      (gv4i*)(uintptr_t)g, (lv4i*)(uintptr_t)(uint32_t)(uintptr_t)lds, 0, 0);
#else
  *(v8h*)lds = *(const v8h*)g;
#endif
}

__device__ __forceinline__ void wait_async0() {
#ifdef HAS_ASYNC
#if __has_builtin(__builtin_amdgcn_s_wait_asynccnt)
  __builtin_amdgcn_s_wait_asynccnt(0);
#else
  asm volatile("s_wait_asynccnt 0x0" ::: "memory");
#endif
#endif
}

// ---------------- one-shot preconversion kernels ----------------

__global__ __launch_bounds__(256) void cvt_f16(const float* __restrict__ in,
                                               _Float16* __restrict__ out) {
  const size_t i = ((size_t)blockIdx.x * 256 + threadIdx.x) * 4;
  float4 f = *(const float4*)(in + i);
  v4h h = {(_Float16)f.x, (_Float16)f.y, (_Float16)f.z, (_Float16)f.w};
  *(v4h*)(out + i) = h;
}

// W f32 [K][N] -> Wt f16 [N][K]
__global__ __launch_bounds__(256) void transpose_f16(const float* __restrict__ W,
                                                     _Float16* __restrict__ Wt) {
  __shared__ _Float16 tile[32][33];
  const int k0 = blockIdx.x * 32, n0 = blockIdx.y * 32;
  const int tc = threadIdx.x & 31, tr = threadIdx.x >> 5;  // 8 rows/pass
#pragma unroll
  for (int i = 0; i < 32; i += 8)
    tile[tr + i][tc] = (_Float16)W[(size_t)(k0 + tr + i) * Dc + n0 + tc];
  __syncthreads();
#pragma unroll
  for (int i = 0; i < 32; i += 8)
    Wt[(size_t)(n0 + tr + i) * Dc + k0 + tc] = tile[tc][tr + i];
}

// ---------------- WMMA GEMM: C = A @ Bt^T + bias ----------------
// A f16 [M][1024]; Bt f16 [N][K] (k-contiguous).  OUTM:
//   0 = f16 [B,H,S,HD] (Q,K) | 1 = f16 [B,H,HD,S] (V^T) | 2 = f32 [M,D]
template <int OUTM>
__global__ __launch_bounds__(256) void gemm_wmma(const _Float16* __restrict__ A,
                                                 const _Float16* __restrict__ Bt,
                                                 const float* __restrict__ bias,
                                                 void* __restrict__ Out) {
  __shared__ _Float16 as[2][BM * LDT];
  __shared__ _Float16 bs[2][BN * LDT];
  const int t = threadIdx.x;
  const int lane = t & 31;
  const int wave = t >> 5;
  const int wm = wave >> 1;  // 0..3 -> 32 rows
  const int wn = wave & 1;   // 0..1 -> 64 cols
  const int gm0 = blockIdx.x * BM;
  const int gn0 = blockIdx.y * BN;

  v8f acc[2][4] = {};

  // stage one (A,B) tile pair: 512 chunks of 16B each, 2 per thread per tile
  auto stage = [&](int buf, int k0) {
#pragma unroll
    for (int j = 0; j < 2; j++) {
      const int c = t + j * 256;
      const int row = c >> 2, part = (c & 3) * 8;
      copy16(&as[buf][row * LDT + part], A + (size_t)(gm0 + row) * Dc + k0 + part);
      copy16(&bs[buf][row * LDT + part], Bt + (size_t)(gn0 + row) * Dc + k0 + part);
    }
  };

  stage(0, 0);
  wait_async0();
  __syncthreads();

  int buf = 0;
  for (int k0 = 0; k0 < Dc; k0 += BK) {
    if (k0 + BK < Dc) stage(buf ^ 1, k0 + BK);  // overlap DMA with WMMA
    v16h am0 = load_frag_a(&as[buf][(wm * 32) * LDT], LDT);
    v16h am1 = load_frag_a(&as[buf][(wm * 32 + 16) * LDT], LDT);
#pragma unroll
    for (int nt = 0; nt < 4; nt++) {
      v16h bf = load_frag_b(&bs[buf][(wn * 64 + nt * 16) * LDT], LDT);
      acc[0][nt] = wmma_f16(am0, bf, acc[0][nt]);
      acc[1][nt] = wmma_f16(am1, bf, acc[1][nt]);
    }
    wait_async0();
    __syncthreads();
    buf ^= 1;
  }

  const int half = lane >> 4;
  const int nlane = lane & 15;
#pragma unroll
  for (int mt = 0; mt < 2; mt++) {
#pragma unroll
    for (int nt = 0; nt < 4; nt++) {
      const int gn = gn0 + wn * 64 + nt * 16 + nlane;
      const float bv = bias[gn];
#pragma unroll
      for (int r = 0; r < 8; r++) {
        const int gm = gm0 + wm * 32 + mt * 16 + r + half * 8;
        const float val = acc[mt][nt][r] + bv;
        if (OUTM == 2) {
          ((float*)Out)[(size_t)gm * Dc + gn] = val;
        } else {
          const int b = gm >> 11, s = gm & (Sc - 1);
          const int h = gn >> 6, hd = gn & (HDc - 1);
          _Float16* O = (_Float16*)Out;
          if (OUTM == 0)
            O[(((size_t)b * Hc + h) * Sc + s) * HDc + hd] = (_Float16)val;
          else
            O[(((size_t)b * Hc + h) * HDc + hd) * Sc + s] = (_Float16)val;
        }
      }
    }
  }
}

// ---------------- flash attention ----------------
// 4 waves/block; wave owns 16 query rows of one (b,h); 128-key blocks.
// Row-sums come from a ones-matrix WMMA (no shuffles); only the row-max
// needs a 4-round butterfly, amortized over 128 keys.
__global__ __launch_bounds__(128) void attn_wmma(const _Float16* __restrict__ Q,
                                                 const _Float16* __restrict__ K,
                                                 const _Float16* __restrict__ Vt,
                                                 _Float16* __restrict__ Aout) {
  __shared__ _Float16 pbuf[4][16 * LDP];
  const int lane = threadIdx.x & 31;
  const int wave = threadIdx.x >> 5;
  const int half = lane >> 4;
  const int nlane = lane & 15;
  const int bh = blockIdx.x;  // b*H + h
  const int b = bh >> 4, h = bh & 15;
  const int q0 = blockIdx.y * 64 + wave * 16;

  const _Float16* qb = Q + ((size_t)bh * Sc + q0) * HDc;
  const v16h hscale = (v16h)(_Float16)0.125f;  // 1/sqrt(64), exact in f16
  const v16h aq0 = load_frag_a(qb, HDc) * hscale;       // hd 0..31
  const v16h aq1 = load_frag_a(qb + 32, HDc) * hscale;  // hd 32..63
  const v16h ones = (v16h)(_Float16)1.0f;

  v8f o[4] = {};
  float mrow[8], lrow[8];
#pragma unroll
  for (int r = 0; r < 8; r++) { mrow[r] = -3.0e30f; lrow[r] = 0.0f; }
  _Float16* pb = &pbuf[wave][0];

  for (int k0 = 0; k0 < Sc; k0 += KB) {
    const _Float16* kb = K + ((size_t)bh * Sc + k0) * HDc;
    v8f s[8];
#pragma unroll
    for (int g = 0; g < 8; g++) {  // 8 groups of 16 keys
      const _Float16* gp = kb + g * 16 * HDc;
      v8f a = {};
      a = wmma_f16(aq0, load_frag_b(gp, HDc), a);
      a = wmma_f16(aq1, load_frag_b(gp + 32, HDc), a);
      s[g] = a;
    }
    // running row-max + rescale of O / l
#pragma unroll
    for (int r = 0; r < 8; r++) {
      float v = s[0][r];
#pragma unroll
      for (int g = 1; g < 8; g++) v = fmaxf(v, s[g][r]);
#pragma unroll
      for (int off = 8; off > 0; off >>= 1) v = fmaxf(v, __shfl_xor(v, off, 32));
      const float mn = fmaxf(mrow[r], v);
      const float al = __expf(mrow[r] - mn);
      mrow[r] = mn;
      lrow[r] *= al;
      o[0][r] *= al; o[1][r] *= al; o[2][r] *= al; o[3][r] *= al;
    }
    // P = exp(s - m) -> LDS (C-layout -> A-layout bounce)
#pragma unroll
    for (int g = 0; g < 8; g++) {
#pragma unroll
      for (int r = 0; r < 8; r++) {
        const float p = __expf(s[g][r] - mrow[r]);
        pb[(r + half * 8) * LDP + g * 16 + nlane] = (_Float16)p;
      }
    }
    asm volatile("s_wait_dscnt 0x0" ::: "memory");  // cross-lane LDS dep
    v16h pf[4];
#pragma unroll
    for (int c = 0; c < 4; c++) pf[c] = load_frag_a(pb + c * 32, LDP);
    // row-sums via ones-matrix WMMA: every lane gets its row's sum
    v8f rs = {};
#pragma unroll
    for (int c = 0; c < 4; c++) rs = wmma_f16(pf[c], ones, rs);
#pragma unroll
    for (int r = 0; r < 8; r++) lrow[r] += rs[r];
    // O += P @ V
    const _Float16* vb = Vt + (size_t)bh * HDc * Sc + k0;
#pragma unroll
    for (int nt = 0; nt < 4; nt++) {
#pragma unroll
      for (int c = 0; c < 4; c++) {
        o[nt] = wmma_f16(pf[c],
                         load_frag_b(vb + (size_t)nt * 16 * Sc + c * 32, Sc),
                         o[nt]);
      }
    }
  }

#pragma unroll
  for (int r = 0; r < 8; r++) {
    const float inv = 1.0f / lrow[r];
    const int q = q0 + r + half * 8;
#pragma unroll
    for (int nt = 0; nt < 4; nt++) {
      Aout[((size_t)b * Sc + q) * Dc + h * HDc + nt * 16 + nlane] =
          (_Float16)(o[nt][r] * inv);
    }
  }
}

extern "C" void kernel_launch(void* const* d_in, const int* in_sizes, int n_in,
                              void* d_out, int out_size, void* d_ws,
                              size_t ws_size, hipStream_t stream) {
  (void)in_sizes; (void)n_in; (void)out_size;
  const float* x  = (const float*)d_in[0];
  const float* Wq = (const float*)d_in[1];
  const float* bq = (const float*)d_in[2];
  const float* Wk = (const float*)d_in[3];
  const float* bk = (const float*)d_in[4];
  const float* Wv = (const float*)d_in[5];
  const float* bv = (const float*)d_in[6];
  const float* Wo = (const float*)d_in[7];
  const float* bo = (const float*)d_in[8];

  const size_t xel = (size_t)Mc * Dc;  // 8M
  const size_t wel = (size_t)Dc * Dc;  // 1M
  if (ws_size < (5 * xel + 4 * wel) * sizeof(_Float16)) return;  // 88MB
  _Float16* Xh  = (_Float16*)d_ws;
  _Float16* WtQ = Xh + xel;
  _Float16* WtK = WtQ + wel;
  _Float16* WtV = WtK + wel;
  _Float16* WtO = WtV + wel;
  _Float16* Qh  = WtO + wel;
  _Float16* Kh  = Qh + xel;
  _Float16* Vt  = Kh + xel;
  _Float16* Ah  = Vt + xel;

  cvt_f16<<<dim3(xel / (256 * 4)), dim3(256), 0, stream>>>(x, Xh);
  dim3 tg(Dc / 32, Dc / 32), tb(256);
  transpose_f16<<<tg, tb, 0, stream>>>(Wq, WtQ);
  transpose_f16<<<tg, tb, 0, stream>>>(Wk, WtK);
  transpose_f16<<<tg, tb, 0, stream>>>(Wv, WtV);
  transpose_f16<<<tg, tb, 0, stream>>>(Wo, WtO);

  dim3 gg(Mc / BM, Dc / BN), gb(256);
  gemm_wmma<0><<<gg, gb, 0, stream>>>(Xh, WtQ, bq, Qh);
  gemm_wmma<0><<<gg, gb, 0, stream>>>(Xh, WtK, bk, Kh);
  gemm_wmma<1><<<gg, gb, 0, stream>>>(Xh, WtV, bv, Vt);

  attn_wmma<<<dim3(Bc * Hc, Sc / 64), dim3(128), 0, stream>>>(Qh, Kh, Vt, Ah);

  gemm_wmma<2><<<gg, gb, 0, stream>>>(Ah, WtO, bo, (float*)d_out);
}
